// layer_30640296690011
// MI455X (gfx1250) — compile-verified, matrix operand order checked by reading, not run on hardware
//
#include <hip/hip_runtime.h>
#include <hip/hip_bf16.h>
#include <math.h>

// Problem dims
#define BB 64
#define TT 65            // T+1 steps
#define EE 512
#define HH 512
#define NG 2048          // 4H gate columns
#define KKD 512
#define MM (TT*BB)       // 4160 rows (t*64+b)
#define NV 32001
#define NVP 32016        // padded to 16
#define NT_OUT 2001      // 32016/16 N tiles for output gemm
#define SCAN_WGS 32

typedef __attribute__((ext_vector_type(16))) __bf16        v16bf;
typedef __attribute__((ext_vector_type(8)))  float         v8f;
typedef __attribute__((ext_vector_type(4)))  unsigned int  u32x4;

union FragU { u32x4 u[2]; v16bf v; };

__device__ __forceinline__ v8f wmma_bf16(v16bf a, v16bf b, v8f c) {
  return __builtin_amdgcn_wmma_f32_16x16x32_bf16(false, a, false, b, (short)0, c,
                                                 false, false);
}

// A fragment: 16x32 bf16 tile from row-major [rows x rowstride] bf16 array.
// lane l: M = l%16 ; K base = (l<16?0:8) ; values K..K+7 then K+16..K+23.
__device__ __forceinline__ v16bf load_frag_a(const __hip_bfloat16* base, int row,
                                             int rowstride, int kbase) {
  FragU f;
  const __hip_bfloat16* p = base + (long)row * rowstride + kbase;
  f.u[0] = *((const u32x4*)p);
  f.u[1] = *((const u32x4*)(p + 16));
  return f.v;
}

// B fragment from pre-packed buffer: block `blk` = (ntile*nKt + ktile); each
// block is 32 lanes * 16 bf16 = 1KB contiguous.
__device__ __forceinline__ v16bf load_frag_b(const __hip_bfloat16* packed, long blk,
                                             int lane) {
  FragU f;
  const u32x4* p = (const u32x4*)(packed + blk * 512 + lane * 16);
  f.u[0] = p[0];
  f.u[1] = p[1];
  return f.v;
}

// ---------------- pack weights into per-lane WMMA B fragment layout ----------
// mode 0/1: gate-interleaved columns (n = 4*unit + gate -> src row gate*512+unit)
// mode 2:   identity columns, pad n >= 32001 with zero (W_out)
__global__ void pack_b_kernel(const float* __restrict__ src,
                              __hip_bfloat16* __restrict__ dst,
                              int mode) {
  int blk = blockIdx.x;            // blk = nt*16 + kt   (nKt = 512/32 = 16)
  int nt  = blk >> 4;
  int kt  = blk & 15;
  for (int idx = threadIdx.x; idx < 512; idx += blockDim.x) {
    int lane = idx >> 4;
    int p    = idx & 15;
    int n    = nt * 16 + (lane & 15);
    int kb   = (lane < 16) ? 0 : 8;
    int k    = kt * 32 + kb + (p < 8 ? p : p + 8);
    float val;
    if (mode == 2) {
      val = (n < NV) ? src[(long)n * KKD + k] : 0.0f;
    } else {
      int unit = n >> 2, gate = n & 3;
      val = src[(long)(gate * HH + unit) * KKD + k];
    }
    dst[(long)blk * 512 + idx] = __float2bfloat16(val);
  }
}

// ---------------- gather: build bf16 input rows (t==0 -> encoded, else emb) --
__global__ void gather_a_kernel(const float* __restrict__ encoded,
                                const int* __restrict__ seq,
                                const float* __restrict__ emb,
                                __hip_bfloat16* __restrict__ Ain) {
  int m = blockIdx.x;              // 0..4159, m = t*64 + b
  int t = m >> 6, b = m & 63;
  const float* srcrow;
  if (t == 0) srcrow = encoded + (long)b * EE;
  else        srcrow = emb + (long)seq[b * 64 + (t - 1)] * EE;
  for (int e = threadIdx.x; e < EE; e += blockDim.x)
    Ain[(long)m * EE + e] = __float2bfloat16(srcrow[e]);
}

// ---------------- init: interleaved bias, padded out-bias, h=0, counter=0 ----
__global__ void init_misc_kernel(const float* __restrict__ b,
                                 const float* __restrict__ b_out,
                                 float* __restrict__ bil,
                                 float* __restrict__ bop,
                                 __hip_bfloat16* __restrict__ h0,
                                 unsigned int* __restrict__ counter) {
  int i = blockIdx.x * blockDim.x + threadIdx.x;
  if (i < NG) { int unit = i >> 2, gate = i & 3; bil[i] = b[gate * HH + unit]; }
  if (i < NVP) bop[i] = (i < NV) ? b_out[i] : 0.0f;
  if (i < 2 * BB * HH) h0[i] = __float2bfloat16(0.0f);
  if (i == 0) *counter = 0u;
}

// ---------------- GEMM1: x_pre = Ain(4160x512) * Bih + bias ------------------
// Wave tile: 4 M-tiles x 2 N-tiles. B pair stays in registers; one A frag
// live at a time (low pressure, no spills).
__global__ void __launch_bounds__(256)
gemm_xpre_kernel(const __hip_bfloat16* __restrict__ Ain,
                 const __hip_bfloat16* __restrict__ Bih,
                 const float* __restrict__ bil,
                 float* __restrict__ xpre) {
  int wave = threadIdx.x >> 5, lane = threadIdx.x & 31;
  int mg = blockIdx.x;                           // 0..64, 64 rows each
  int ntbase = (blockIdx.y * 8 + wave) * 2;      // blockIdx.y in [0,8)
  int kb = (lane < 16) ? 0 : 8;
  int rlo = lane & 15;
  const __hip_bfloat16* b0 = Bih + (long)ntbase * 16 * 512;
  const __hip_bfloat16* b1 = Bih + (long)(ntbase + 1) * 16 * 512;
  v8f acc[2][4];
  for (int j = 0; j < 2; j++)
    for (int mi = 0; mi < 4; mi++)
      for (int r = 0; r < 8; r++) acc[j][mi][r] = 0.0f;
  for (int kt = 0; kt < 16; kt++) {
    if (kt < 15) {
      __builtin_prefetch(b0 + (long)(kt + 1) * 512 + lane * 16, 0, 1);
      __builtin_prefetch(b1 + (long)(kt + 1) * 512 + lane * 16, 0, 1);
    }
    v16bf bf0 = load_frag_b(b0, kt, lane);
    v16bf bf1 = load_frag_b(b1, kt, lane);
    for (int mi = 0; mi < 4; mi++) {
      v16bf a = load_frag_a(Ain, mg * 64 + mi * 16 + rlo, KKD, kt * 32 + kb);
      acc[0][mi] = wmma_bf16(a, bf0, acc[0][mi]);
      acc[1][mi] = wmma_bf16(a, bf1, acc[1][mi]);
    }
  }
  for (int j = 0; j < 2; j++) {
    int n = (ntbase + j) * 16 + rlo;
    float bv = bil[n];
    for (int mi = 0; mi < 4; mi++) {
      int m0 = mg * 64 + mi * 16 + ((lane < 16) ? 0 : 8);
      for (int r = 0; r < 8; r++)
        xpre[(long)(m0 + r) * NG + n] = acc[j][mi][r] + bv;
    }
  }
}

// ---------------- persistent LSTM scan (32 WGs, grid barrier per step) -------
__global__ void lstm_scan_kernel(const float* __restrict__ xpre,
                                 const __hip_bfloat16* __restrict__ Bhh,
                                 __hip_bfloat16* __restrict__ hbuf,   // 2 x 64 x 512
                                 __hip_bfloat16* __restrict__ hsb,    // 4160 x 512
                                 unsigned int* __restrict__ counter) {
  int wave = threadIdx.x >> 5, lane = threadIdx.x & 31;
  int g = blockIdx.x;                    // owns gate cols [g*64, g*64+64)
  int kb = (lane < 16) ? 0 : 8;
  int qb = lane & ~3;
  float creg[2][8];
  for (int s = 0; s < 2; s++)
    for (int r = 0; r < 8; r++) creg[s][r] = 0.0f;

  for (int t = 0; t < TT; t++) {
    const __hip_bfloat16* hr = hbuf + (long)(t & 1) * BB * HH;
    __hip_bfloat16*       hw = hbuf + (long)((t + 1) & 1) * BB * HH;
    for (int s = 0; s < 2; s++) {
      int ti = wave * 2 + s;             // 16 tiles: 4 mtiles x 4 local ntiles
      int mt = ti >> 2, ntl = ti & 3;
      int nt = g * 4 + ntl;
      int row = mt * 16 + (lane & 15);
      v8f acc;
      for (int r = 0; r < 8; r++) acc[r] = 0.0f;
      for (int kt = 0; kt < 16; kt++) {
        v16bf a  = load_frag_a(hr, row, HH, kt * 32 + kb);
        v16bf bf = load_frag_b(Bhh, (long)nt * 16 + kt, lane);
        acc = wmma_bf16(a, bf, acc);
      }
      int n = nt * 16 + (lane & 15);     // global gate column (interleaved)
      int gate = n & 3, unit = n >> 2;
      int m0 = mt * 16 + ((lane < 16) ? 0 : 8);
      for (int r = 0; r < 8; r++) {
        int bidx = m0 + r;               // batch
        float d = acc[r] + xpre[(long)(t * BB + bidx) * NG + n];
        float aact = (gate == 2) ? tanhf(d) : 1.0f / (1.0f + __expf(-d));
        float iv = __shfl(aact, qb + 0, 32);
        float fv = __shfl(aact, qb + 1, 32);
        float gv = __shfl(aact, qb + 2, 32);
        float ov = __shfl(aact, qb + 3, 32);
        float cn = fv * creg[s][r] + iv * gv;
        creg[s][r] = cn;
        float hv = ov * tanhf(cn);
        if ((lane & 3) == 0) {
          __hip_bfloat16 hb = __float2bfloat16(hv);
          hw[(long)bidx * HH + unit] = hb;
          hsb[(long)(t * BB + bidx) * HH + unit] = hb;
        }
      }
    }
    // grid-wide barrier (monotonic counter; 32 resident blocks)
    __syncthreads();
    if (threadIdx.x == 0) {
      __threadfence();
      __hip_atomic_fetch_add(counter, 1u, __ATOMIC_ACQ_REL, __HIP_MEMORY_SCOPE_AGENT);
      unsigned target = (unsigned)(t + 1) * SCAN_WGS;
      while (__hip_atomic_load(counter, __ATOMIC_ACQUIRE, __HIP_MEMORY_SCOPE_AGENT) < target)
        __builtin_amdgcn_s_sleep(1);
    }
    __syncthreads();
  }
}

// ---------------- GEMM3: logits = hsb(4160x512) * Bout + b_out -> d_out ------
// Wave tile: 4 M-tiles x 2 N-tiles. B pair resident, one A frag live.
__global__ void __launch_bounds__(256)
gemm_out_kernel(const __hip_bfloat16* __restrict__ hsb,
                const __hip_bfloat16* __restrict__ Bout,
                const float* __restrict__ bop,
                float* __restrict__ out) {
  int wave = threadIdx.x >> 5, lane = threadIdx.x & 31;
  int mg = blockIdx.x;                               // 0..64, 64 rows each
  int ntbase = (blockIdx.y * 8 + wave) * 2;          // blockIdx.y in [0,126)
  if (ntbase >= NT_OUT) return;                      // wave-uniform
  int haveN2 = (ntbase + 1 < NT_OUT);                // wave-uniform
  int kb = (lane < 16) ? 0 : 8;
  int rlo = lane & 15;
  const __hip_bfloat16* b0 = Bout + (long)ntbase * 16 * 512;
  const __hip_bfloat16* b1 = Bout + (long)(ntbase + (haveN2 ? 1 : 0)) * 16 * 512;
  v8f acc[2][4];
  for (int j = 0; j < 2; j++)
    for (int mi = 0; mi < 4; mi++)
      for (int r = 0; r < 8; r++) acc[j][mi][r] = 0.0f;
  for (int kt = 0; kt < 16; kt++) {
    if (kt < 15) {
      __builtin_prefetch(b0 + (long)(kt + 1) * 512 + lane * 16, 0, 1);
      __builtin_prefetch(b1 + (long)(kt + 1) * 512 + lane * 16, 0, 1);
    }
    v16bf bf0 = load_frag_b(b0, kt, lane);
    v16bf bf1 = load_frag_b(b1, kt, lane);
    for (int mi = 0; mi < 4; mi++) {
      v16bf a = load_frag_a(hsb, mg * 64 + mi * 16 + rlo, HH, kt * 32 + kb);
      acc[0][mi] = wmma_bf16(a, bf0, acc[0][mi]);
      acc[1][mi] = wmma_bf16(a, bf1, acc[1][mi]);
    }
  }
  int n2 = haveN2 ? 2 : 1;
  for (int j = 0; j < n2; j++) {
    int n = (ntbase + j) * 16 + rlo;
    if (n < NV) {
      float bv = bop[n];
      for (int mi = 0; mi < 4; mi++) {
        int m0 = mg * 64 + mi * 16 + ((lane < 16) ? 0 : 8);
        for (int r = 0; r < 8; r++) {
          int m = m0 + r;
          int t = m >> 6, b = m & 63;                // m = t*64+b
          out[(long)(b * TT + t) * NV + n] = acc[j][mi][r] + bv;
        }
      }
    }
  }
}

// ---------------- row-wise log-softmax, in place over d_out ------------------
__global__ void logsoftmax_kernel(float* __restrict__ out) {
  float* row = out + (long)blockIdx.x * NV;
  float mx = -INFINITY, sm = 0.0f;
  for (int i = threadIdx.x; i < NV; i += 256) {
    float x = row[i];
    if (x > mx) { sm = sm * __expf(mx - x) + 1.0f; mx = x; }
    else        { sm += __expf(x - mx); }
  }
  __shared__ float smx[256], ssm[256];
  smx[threadIdx.x] = mx; ssm[threadIdx.x] = sm;
  __syncthreads();
  for (int off = 128; off > 0; off >>= 1) {
    if (threadIdx.x < off) {
      float m1 = smx[threadIdx.x], s1 = ssm[threadIdx.x];
      float m2 = smx[threadIdx.x + off], s2 = ssm[threadIdx.x + off];
      float mm = fmaxf(m1, m2);
      smx[threadIdx.x] = mm;
      ssm[threadIdx.x] = s1 * __expf(m1 - mm) + s2 * __expf(m2 - mm);
    }
    __syncthreads();
  }
  float lse = smx[0] + __logf(ssm[0]);
  __syncthreads();
  for (int i = threadIdx.x; i < NV; i += 256) row[i] -= lse;
}

// ---------------- host orchestration ----------------------------------------
extern "C" void kernel_launch(void* const* d_in, const int* in_sizes, int n_in,
                              void* d_out, int out_size, void* d_ws, size_t ws_size,
                              hipStream_t stream) {
  const float* encoded = (const float*)d_in[0];
  const int*   seq     = (const int*)  d_in[1];
  // d_in[2] = lengths (unused by reference math)
  const float* emb     = (const float*)d_in[3];
  const float* W_ih    = (const float*)d_in[4];
  const float* W_hh    = (const float*)d_in[5];
  const float* b       = (const float*)d_in[6];
  const float* W_out   = (const float*)d_in[7];
  const float* b_out   = (const float*)d_in[8];
  float* out = (float*)d_out;

  char* ws = (char*)d_ws;
  size_t off = 0;
  auto take = [&](size_t bytes) { char* p = ws + off; off = (off + bytes + 255) & ~(size_t)255; return p; };
  __hip_bfloat16* Bih  = (__hip_bfloat16*)take((size_t)NG  * KKD * 2);
  __hip_bfloat16* Bhh  = (__hip_bfloat16*)take((size_t)NG  * KKD * 2);
  __hip_bfloat16* Bout = (__hip_bfloat16*)take((size_t)NVP * KKD * 2);
  __hip_bfloat16* Ain  = (__hip_bfloat16*)take((size_t)MM  * EE  * 2);
  float*          xpre = (float*)         take((size_t)MM  * NG  * 4);
  __hip_bfloat16* hsb  = (__hip_bfloat16*)take((size_t)MM  * HH  * 2);
  __hip_bfloat16* hbuf = (__hip_bfloat16*)take((size_t)2 * BB * HH * 2);
  float*          bil  = (float*)         take((size_t)NG  * 4);
  float*          bop  = (float*)         take((size_t)NVP * 4);
  unsigned int*   ctr  = (unsigned int*)  take(256);
  (void)ws_size; (void)in_sizes; (void)n_in; (void)out_size;

  pack_b_kernel<<<dim3(128 * 16), dim3(256), 0, stream>>>(W_ih,  Bih,  0);
  pack_b_kernel<<<dim3(128 * 16), dim3(256), 0, stream>>>(W_hh,  Bhh,  1);
  pack_b_kernel<<<dim3(NT_OUT * 16), dim3(256), 0, stream>>>(W_out, Bout, 2);
  gather_a_kernel<<<dim3(MM), dim3(256), 0, stream>>>(encoded, seq, emb, Ain);
  init_misc_kernel<<<dim3(256), dim3(256), 0, stream>>>(b, b_out, bil, bop, hbuf, ctr);

  gemm_xpre_kernel<<<dim3(MM / 64, 8), dim3(256), 0, stream>>>(Ain, Bih, bil, xpre);
  lstm_scan_kernel<<<dim3(SCAN_WGS), dim3(256), 0, stream>>>(xpre, Bhh, hbuf, hsb, ctr);
  gemm_out_kernel<<<dim3(MM / 64, 126), dim3(256), 0, stream>>>(hsb, Bout, bop, out);
  logsoftmax_kernel<<<dim3(MM), dim3(256), 0, stream>>>(out);
}
